// Block_70944269795510
// MI455X (gfx1250) — matrix operand order repaired and checked by reading, not compiled, not run on hardware
//
#include <hip/hip_runtime.h>
#include <math.h>

// ---- problem constants ----
#define BB    16
#define CC    256
#define HH    56
#define WWID  56
#define HWSZ  (HH*WWID)      // 3136
#define KKER  7
#define GG    16
#define GCH   16
#define REDC  64
#define SPANC (KKER*KKER*GG) // 784
#define EPSV  1e-5f

typedef __attribute__((ext_vector_type(16))) _Float16 v16h;
typedef __attribute__((ext_vector_type(8)))  _Float16 v8h;
typedef __attribute__((ext_vector_type(8)))  float    v8f;

// ---- coef table layout (floats) ----
// [0]    s1[256]   [256] h1[256]   (bn1, b1 folded)
// [512]  s2[256]   [768] h2[256]   (bn2)
// [1024] s3[256]   [1280] h3[256]  (bn3, b3 folded)
// [1536] sr[64]    [1600] hr[64]   (red bn, red_b folded)
// [1664] ss[784]   [2448] hs[784]  (span: s=1, h=span_b)

__global__ void coef_kernel(const float* __restrict__ b1,
                            const float* __restrict__ g1, const float* __restrict__ bt1,
                            const float* __restrict__ m1, const float* __restrict__ v1,
                            const float* __restrict__ rb,
                            const float* __restrict__ rg, const float* __restrict__ rbt,
                            const float* __restrict__ rm, const float* __restrict__ rv,
                            const float* __restrict__ g2, const float* __restrict__ bt2,
                            const float* __restrict__ m2, const float* __restrict__ v2,
                            const float* __restrict__ b3,
                            const float* __restrict__ g3, const float* __restrict__ bt3,
                            const float* __restrict__ m3, const float* __restrict__ v3,
                            const float* __restrict__ sb,
                            float* __restrict__ coef) {
    int t = blockIdx.x * blockDim.x + threadIdx.x;   // 0..1023
    if (t < CC) {
        float s = g1[t] * rsqrtf(v1[t] + EPSV);
        coef[t]        = s;
        coef[256 + t]  = bt1[t] - m1[t] * s + b1[t] * s;
        float s2 = g2[t] * rsqrtf(v2[t] + EPSV);
        coef[512 + t]  = s2;
        coef[768 + t]  = bt2[t] - m2[t] * s2;
        float s3 = g3[t] * rsqrtf(v3[t] + EPSV);
        coef[1024 + t] = s3;
        coef[1280 + t] = bt3[t] - m3[t] * s3 + b3[t] * s3;
    }
    if (t < REDC) {
        float sr = rg[t] * rsqrtf(rv[t] + EPSV);
        coef[1536 + t] = sr;
        coef[1600 + t] = rbt[t] - rm[t] * sr + rb[t] * sr;
    }
    if (t < SPANC) {
        coef[1664 + t] = 1.0f;
        coef[2448 + t] = sb[t];
    }
}

// ---- pre-swizzle W (Ndim,Kdim) f32 -> f16 A-fragment layout ----
// element idx = (((nt*(Kdim/32) + kb)*32) + lane)*16 + e
// lane: m = nt*16 + (lane&15), half = lane>>4
// k = kb*32 + half*16 + e
__global__ void pack_w(const float* __restrict__ W, _Float16* __restrict__ out,
                       int Kdim, int total) {
    int idx = blockIdx.x * blockDim.x + threadIdx.x;
    if (idx >= total) return;
    int e    = idx & 15;
    int l    = (idx >> 4) & 31;
    int rest = idx >> 9;
    int nkb  = Kdim >> 5;
    int kb   = rest % nkb;
    int nt   = rest / nkb;
    int m    = nt * 16 + (l & 15);
    int half = l >> 4;
    int k    = kb * 32 + half * 16 + e;
    out[idx] = (_Float16)W[(long)m * Kdim + k];
}

// ---- pack x (B,C,H,W) f32 -> xh [p][c] f16 via LDS tile transpose ----
// grid(49, 16, 16)  block(256): tile = 64 pixels x 16 channels
__global__ void pack_x(const float* __restrict__ x, _Float16* __restrict__ xh) {
    __shared__ _Float16 T[16][65];
    int t  = threadIdx.x;
    int pt = blockIdx.x, ct = blockIdx.y, b = blockIdx.z;
    int hw0 = pt * 64, c0 = ct * 16;
    const float* src = x + (long)b * CC * HWSZ;
    int px = t & 63, cl = t >> 6;
#pragma unroll
    for (int r = 0; r < 4; r++) {
        int c = cl * 4 + r;
        T[c][px] = (_Float16)src[(long)(c0 + c) * HWSZ + hw0 + px];
    }
    __syncthreads();
    int c = t & 15, p2 = t >> 4;
#pragma unroll
    for (int r = 0; r < 4; r++) {
        int pp = p2 + r * 16;
        xh[((long)b * HWSZ + hw0 + pp) * CC + c0 + c] = T[c][pp];
    }
}

// ---- WMMA GEMM: one wave computes a 16(M) x 64(P) strip, K steps of 32 ----
// A from pre-swizzled f16 fragments (no cvt in loop). OP: 0 none, 1 relu, 2 tanh
template <int OP>
__global__ __launch_bounds__(32) void gemm_f16(
        const _Float16* __restrict__ Wswz, // swizzled (Ndim,Kdim) f16 fragments
        const _Float16* __restrict__ act,  // [p][Kdim] f16
        _Float16*       __restrict__ outh, // [p][Ndim] f16
        const float*   __restrict__ scale,
        const float*   __restrict__ shift,
        int Kdim, int Ndim) {
    int l    = threadIdx.x;      // 0..31
    int n    = l & 15;
    int half = l >> 4;
    int pt = blockIdx.x, nt = blockIdx.y, b = blockIdx.z;
    long pbase = (long)b * HWSZ + pt * 64;
    int  n0 = nt * 16;
    int  nkb = Kdim >> 5;

    const _Float16* abase = Wswz + ((long)nt * nkb * 32 + l) * 16;
    const _Float16* bcol0 = act + (pbase + n) * Kdim;

    v8f acc[4] = {{}, {}, {}, {}};
    for (int kb = 0; kb < nkb; kb++) {
        v8h alo = *(const v8h*)(abase + (long)kb * 512);
        v8h ahi = *(const v8h*)(abase + (long)kb * 512 + 8);
        v16h a;
#pragma unroll
        for (int i = 0; i < 8; i++) { a[i] = alo[i]; a[i + 8] = ahi[i]; }

        int ko = kb * 32 + half * 16;
#pragma unroll
        for (int tt = 0; tt < 4; tt++) {
            const _Float16* bc = bcol0 + (long)tt * 16 * Kdim + ko;
            v8h blo = *(const v8h*)bc;
            v8h bhi = *(const v8h*)(bc + 8);
            v16h bm;
#pragma unroll
            for (int i = 0; i < 8; i++) { bm[i] = blo[i]; bm[i + 8] = bhi[i]; }
            acc[tt] = __builtin_amdgcn_wmma_f32_16x16x32_f16(false, a, false, bm,
                                                             (short)0, acc[tt], false, false);
        }
    }
    // D fragment: lane n=l&15 ; VGPR j -> M = half*8 + j  (same m across the 4 tiles)
    int mbase = n0 + half * 8;
    v8h o[4];
#pragma unroll
    for (int j = 0; j < 8; j++) {
        float sc = scale[mbase + j];
        float sh = shift[mbase + j];
#pragma unroll
        for (int tt = 0; tt < 4; tt++) {
            float v = acc[tt][j] * sc + sh;
            if (OP == 1) v = v > 0.f ? v : 0.f;
            if (OP == 2) v = tanhf(v);
            o[tt][j] = (_Float16)v;
        }
    }
#pragma unroll
    for (int tt = 0; tt < 4; tt++)
        *(v8h*)(outh + (pbase + tt * 16 + n) * Ndim + mbase) = o[tt];
}

// ---- involution gather + bn2 + tanh -> th (f16) ----
// grid(196, G=16, B=16) block(256): thread = (c = t&15, local pixel = t>>4)
__global__ void involution_kernel(const _Float16* __restrict__ yh,   // [p][256]
                                  const _Float16* __restrict__ wdh,  // [p][784]
                                  _Float16*       __restrict__ th,   // [p][256]
                                  const float*   __restrict__ s2,
                                  const float*   __restrict__ h2) {
    __shared__ _Float16 WS[16][49];   // dynamic kernels for the block's 16 pixels
    int t  = threadIdx.x;
    int b  = blockIdx.z, g = blockIdx.y;
    int hw0 = blockIdx.x * 16;
    long pbase = (long)b * HWSZ + hw0;

    for (int i = t; i < 16 * 49; i += 256) {
        int lp = i / 49, k = i - lp * 49;
        WS[lp][k] = wdh[(pbase + lp) * SPANC + g * 49 + k];
    }
    __syncthreads();

    int c  = t & 15, lp = t >> 4;
    int hw = hw0 + lp;
    int h  = hw / WWID, w = hw - h * WWID;
    int ch = g * GCH + c;

    float acc = 0.f;
#pragma unroll
    for (int di = -3; di <= 3; di++) {
        int hh = h + di;
        if ((unsigned)hh >= (unsigned)HH) continue;
#pragma unroll
        for (int dj = -3; dj <= 3; dj++) {
            int wn = w + dj;
            if ((unsigned)wn >= (unsigned)WWID) continue;
            float wv = (float)WS[lp][(di + 3) * 7 + (dj + 3)];
            long pn = (long)b * HWSZ + hh * WWID + wn;
            float yv = (float)yh[pn * CC + ch];
            acc = fmaf(wv, yv, acc);
        }
    }
    float v = tanhf(acc * s2[ch] + h2[ch]);
    th[(pbase + lp) * CC + ch] = (_Float16)v;
}

// ---- final GEMM + bn3 + fp32 skip-add, store (B,C,H,W) f32; 16x64 strip/wave ----
__global__ __launch_bounds__(32) void gemm_out(
        const _Float16* __restrict__ Wswz, // swizzled (256,256) f16 fragments
        const _Float16* __restrict__ act,  // th [p][256]
        const float*   __restrict__ x,     // (B,C,H,W) f32
        float*         __restrict__ out,   // (B,C,H,W) f32
        const float*   __restrict__ s3,
        const float*   __restrict__ h3) {
    int l = threadIdx.x;
    int n = l & 15, half = l >> 4;
    int pt = blockIdx.x, nt = blockIdx.y, b = blockIdx.z;
    int hwb = pt * 64;
    long pbase = (long)b * HWSZ + hwb;
    int n0 = nt * 16;
    const int nkb = CC >> 5;

    const _Float16* abase = Wswz + ((long)nt * nkb * 32 + l) * 16;
    const _Float16* bcol0 = act + (pbase + n) * CC;

    v8f acc[4] = {{}, {}, {}, {}};
    for (int kb = 0; kb < nkb; kb++) {
        v8h alo = *(const v8h*)(abase + (long)kb * 512);
        v8h ahi = *(const v8h*)(abase + (long)kb * 512 + 8);
        v16h a;
#pragma unroll
        for (int i = 0; i < 8; i++) { a[i] = alo[i]; a[i + 8] = ahi[i]; }
        int ko = kb * 32 + half * 16;
#pragma unroll
        for (int tt = 0; tt < 4; tt++) {
            const _Float16* bc = bcol0 + (long)tt * 16 * CC + ko;
            v8h blo = *(const v8h*)bc;
            v8h bhi = *(const v8h*)(bc + 8);
            v16h bm;
#pragma unroll
            for (int i = 0; i < 8; i++) { bm[i] = blo[i]; bm[i + 8] = bhi[i]; }
            acc[tt] = __builtin_amdgcn_wmma_f32_16x16x32_f16(false, a, false, bm,
                                                             (short)0, acc[tt], false, false);
        }
    }
    int mbase = n0 + half * 8;
#pragma unroll
    for (int j = 0; j < 8; j++) {
        int m = mbase + j;
        float sc = s3[m];
        float sh = h3[m];
        long orow = ((long)b * CC + m) * HWSZ + hwb + n;
#pragma unroll
        for (int tt = 0; tt < 4; tt++) {
            long oi = orow + tt * 16;       // 16 lanes -> 64B contiguous per (j,tt)
            out[oi] = acc[tt][j] * sc + sh + x[oi];
        }
    }
}

extern "C" void kernel_launch(void* const* d_in, const int* in_sizes, int n_in,
                              void* d_out, int out_size, void* d_ws, size_t ws_size,
                              hipStream_t stream) {
    const float* x      = (const float*)d_in[0];
    const float* w1     = (const float*)d_in[1];
    const float* b1     = (const float*)d_in[2];
    const float* bn1_g  = (const float*)d_in[3];
    const float* bn1_b  = (const float*)d_in[4];
    const float* bn1_m  = (const float*)d_in[5];
    const float* bn1_v  = (const float*)d_in[6];
    const float* red_w  = (const float*)d_in[7];
    const float* red_b  = (const float*)d_in[8];
    const float* rbn_g  = (const float*)d_in[9];
    const float* rbn_b  = (const float*)d_in[10];
    const float* rbn_m  = (const float*)d_in[11];
    const float* rbn_v  = (const float*)d_in[12];
    const float* span_w = (const float*)d_in[13];
    const float* span_b = (const float*)d_in[14];
    const float* bn2_g  = (const float*)d_in[15];
    const float* bn2_b  = (const float*)d_in[16];
    const float* bn2_m  = (const float*)d_in[17];
    const float* bn2_v  = (const float*)d_in[18];
    const float* w3     = (const float*)d_in[19];
    const float* b3     = (const float*)d_in[20];
    const float* bn3_g  = (const float*)d_in[21];
    const float* bn3_b  = (const float*)d_in[22];
    const float* bn3_m  = (const float*)d_in[23];
    const float* bn3_v  = (const float*)d_in[24];
    float* out = (float*)d_out;

    // workspace carve-up (~163 MB)
    char* ws = (char*)d_ws;
    const long PTOT  = (long)BB * HWSZ;                  // 50176 pixels
    float*    coef = (float*)ws;                         size_t off = 16384;
    _Float16* wh1  = (_Float16*)(ws + off);              off += (size_t)CC * CC * 2;
    _Float16* whr  = (_Float16*)(ws + off);              off += (size_t)REDC * CC * 2;
    _Float16* whs  = (_Float16*)(ws + off);              off += (size_t)SPANC * REDC * 2;
    _Float16* wh3  = (_Float16*)(ws + off);              off += (size_t)CC * CC * 2;
    _Float16* xh   = (_Float16*)(ws + off);              off += PTOT * CC    * 2;
    _Float16* yh   = (_Float16*)(ws + off);              off += PTOT * CC    * 2;
    _Float16* rh   = (_Float16*)(ws + off);              off += PTOT * REDC  * 2;
    _Float16* wdh  = (_Float16*)(ws + off);              off += PTOT * SPANC * 2;
    _Float16* th   = (_Float16*)(ws + off);              off += PTOT * CC    * 2;
    (void)ws_size;

    float* s1 = coef,        *h1 = coef + 256;
    float* s2 = coef + 512,  *h2 = coef + 768;
    float* s3 = coef + 1024, *h3 = coef + 1280;
    float* sr = coef + 1536, *hr = coef + 1600;
    float* ss = coef + 1664, *hs = coef + 2448;

    // 1) fold BN coefficients
    coef_kernel<<<4, 256, 0, stream>>>(b1, bn1_g, bn1_b, bn1_m, bn1_v,
                                       red_b, rbn_g, rbn_b, rbn_m, rbn_v,
                                       bn2_g, bn2_b, bn2_m, bn2_v,
                                       b3, bn3_g, bn3_b, bn3_m, bn3_v,
                                       span_b, coef);
    // 2) swizzle weights into f16 A-fragment layout
    pack_w<<<(CC * CC + 255) / 256, 256, 0, stream>>>(w1, wh1, CC, CC * CC);
    pack_w<<<(REDC * CC + 255) / 256, 256, 0, stream>>>(red_w, whr, CC, REDC * CC);
    pack_w<<<(SPANC * REDC + 255) / 256, 256, 0, stream>>>(span_w, whs, REDC, SPANC * REDC);
    pack_w<<<(CC * CC + 255) / 256, 256, 0, stream>>>(w3, wh3, CC, CC * CC);
    // 3) pack x -> f16 [p][c]
    pack_x<<<dim3(HWSZ / 64, CC / 16, BB), 256, 0, stream>>>(x, xh);
    // 4) y = tanh(bn1(W1 @ x))
    gemm_f16<2><<<dim3(HWSZ / 64, CC / 16, BB), 32, 0, stream>>>(
        wh1, xh, yh, s1, h1, CC, CC);
    // 5) r = relu(bn_red(Wr @ y))
    gemm_f16<1><<<dim3(HWSZ / 64, REDC / 16, BB), 32, 0, stream>>>(
        whr, yh, rh, sr, hr, CC, REDC);
    // 6) wd = Ws @ r + span_b
    gemm_f16<0><<<dim3(HWSZ / 64, SPANC / 16, BB), 32, 0, stream>>>(
        whs, rh, wdh, ss, hs, REDC, SPANC);
    // 7) involution + bn2 + tanh
    involution_kernel<<<dim3(HWSZ / 16, GG, BB), 256, 0, stream>>>(yh, wdh, th, s2, h2);
    // 8) out = bn3(W3 @ t) + x
    gemm_out<<<dim3(HWSZ / 64, CC / 16, BB), 32, 0, stream>>>(w3 ? wh3 : wh3, th, x, out, s3, h3);
}
